// Weighted_QKVAttention_28862180229257
// MI455X (gfx1250) — matrix operand order repaired and checked by reading, not compiled, not run on hardware
//
#include <hip/hip_runtime.h>
#include <hip/hip_bf16.h>

typedef __attribute__((ext_vector_type(16))) _Float16 v16h;
typedef __attribute__((ext_vector_type(8)))  float    v8f;

#define T_LEN    2048
#define CH       64
#define HEADS    8
#define NBH      16     // bs * HEADS
#define BLOCK_T  64     // t rows per block
#define WAVE_T   16     // t rows per wave
#define S_TILE   32     // s columns per iteration
#define NWAVES   4
#define NTHREADS 128
#define LPAD     2      // pad rows to odd dword count -> conflict-free LDS banking

__device__ __forceinline__ v8f wmma_f16(v16h a, v16h b, v8f c) {
    // (neg_a, A, neg_b, B, c_mod, C, reuse_a, reuse_b)
    return __builtin_amdgcn_wmma_f32_16x16x32_f16(false, a, false, b, (short)0, c,
                                                  false, false);
}

// B-fragment (32x16 f16): lane holds column n = lane%16; VGPR v element e holds
// row k = 16*(lane/16) + 2*v + e.  klds is [s][c] (c contiguous) so for the QK
// gemm (K-dim = c) each pair is one contiguous b32 read.
__device__ __forceinline__ v16h load_kb(const _Float16 (*kl)[CH + LPAD],
                                        int sl, int cbase, int g) {
    v16h r;
#pragma unroll
    for (int v = 0; v < 8; ++v) {
        int kc = cbase + 16 * g + 2 * v;
        r[2 * v]     = kl[sl][kc];
        r[2 * v + 1] = kl[sl][kc + 1];
    }
    return r;
}

// vlds is [c][s] (s contiguous) so for the PV gemm (K-dim = s) each pair is one
// contiguous b32 read.
__device__ __forceinline__ v16h load_vb(const _Float16 (*vl)[S_TILE + LPAD],
                                        int ct, int m, int g) {
    v16h r;
    int crow = ct * 16 + m;
#pragma unroll
    for (int v = 0; v < 8; ++v) {
        int ks = 16 * g + 2 * v;
        r[2 * v]     = vl[crow][ks];
        r[2 * v + 1] = vl[crow][ks + 1];
    }
    return r;
}

// A-fragment (16x32 f16): lane holds row m = lane%16; VGPR v element e holds
// k = 16*g + 2*v + e, g = lane/16.
__device__ __forceinline__ int a_kbase(int v, int g) {
    return 16 * g + 2 * v;
}

__global__ __launch_bounds__(NTHREADS)
void wqkv_attn_kernel(const float* __restrict__ qkv,
                      const float* __restrict__ W,
                      float* __restrict__ out) {
    __shared__ _Float16 qlds[BLOCK_T][CH + LPAD];              // [t_local][c]
    __shared__ _Float16 klds[2][S_TILE][CH + LPAD];            // [s_local][c]
    __shared__ _Float16 vlds[2][CH][S_TILE + LPAD];            // [c][s_local]
    __shared__ _Float16 plds[NWAVES][WAVE_T][S_TILE + LPAD];   // [m][s_local]

    const int tid  = threadIdx.x;
    const int wave = tid >> 5;
    const int lane = tid & 31;
    const int m    = lane & 15;   // row/col within 16x16 tile
    const int g    = lane >> 4;   // half-wave

    const int b   = blockIdx.y;   // batch-head 0..15
    const int bsi = b / HEADS;
    const int h   = b % HEADS;
    const int t0b = blockIdx.x * BLOCK_T;
    const int t0  = t0b + wave * WAVE_T;

    const float scale = 0.35355339059327373f;  // 64^-0.25

    const float* __restrict__ qb = qkv + (size_t)bsi * (3 * HEADS * CH) * T_LEN
                                       + (size_t)(h * CH) * T_LEN;
    const float* __restrict__ kb = qb + (size_t)(HEADS * CH) * T_LEN;
    const float* __restrict__ vb = qb + (size_t)(2 * HEADS * CH) * T_LEN;
    const float* __restrict__ Wb = W + (size_t)b * T_LEN * T_LEN;
    float* __restrict__ ob = out + (size_t)bsi * (HEADS * CH) * T_LEN
                                 + (size_t)(h * CH) * T_LEN;

    // ---- stage Q (BLOCK_T x CH) -> LDS [t][c], scaled + f16
    for (int idx = tid; idx < BLOCK_T * CH; idx += NTHREADS) {
        int c  = idx >> 6;            // qkv row; coalesced over tl
        int tl = idx & (BLOCK_T - 1);
        qlds[tl][c] = (_Float16)(qb[(size_t)c * T_LEN + t0b + tl] * scale);
    }
    // ---- stage first K/V tile (buffer 0)
    for (int idx = tid; idx < CH * S_TILE; idx += NTHREADS) {
        int c = idx >> 5, sl = idx & 31;
        klds[0][sl][c] = (_Float16)(kb[(size_t)c * T_LEN + sl] * scale);
        vlds[0][c][sl] = (_Float16)(vb[(size_t)c * T_LEN + sl]);
    }
    __syncthreads();

    // ---- Q A-fragments, loaded once (K-dim = ch: c 0..31 and 32..63)
    v16h qA0, qA1;
#pragma unroll
    for (int v = 0; v < 8; ++v) {
        int k0 = a_kbase(v, g);
        int tl = wave * WAVE_T + m;
        qA0[2 * v]     = qlds[tl][k0];
        qA0[2 * v + 1] = qlds[tl][k0 + 1];
        qA1[2 * v]     = qlds[tl][32 + k0];
        qA1[2 * v + 1] = qlds[tl][32 + k0 + 1];
    }

    v8f O0 = {}, O1 = {}, O2 = {}, O3 = {};
    float Mrow[8], Lrow[8];
#pragma unroll
    for (int r = 0; r < 8; ++r) { Mrow[r] = -__builtin_huge_valf(); Lrow[r] = 0.0f; }

    const int NIT = T_LEN / S_TILE;  // 64
    for (int it = 0; it < NIT; ++it) {
        const int cur = it & 1;
        const int s0  = it * S_TILE;

        // ---- stage next K/V tile into the other buffer (overlaps compute)
        if (it + 1 < NIT) {
            const int s1  = s0 + S_TILE;
            const int nxt = cur ^ 1;
            for (int idx = tid; idx < CH * S_TILE; idx += NTHREADS) {
                int c = idx >> 5, sl = idx & 31;
                klds[nxt][sl][c] = (_Float16)(kb[(size_t)c * T_LEN + s1 + sl] * scale);
                vlds[nxt][c][sl] = (_Float16)(vb[(size_t)c * T_LEN + s1 + sl]);
            }
            // prefetch next Weight_Mat tile rows (global_prefetch_b8)
            __builtin_prefetch(&Wb[(size_t)(t0 + m) * T_LEN + s1 + g * 16], 0, 1);
            __builtin_prefetch(&Wb[(size_t)(t0 + 8 + m) * T_LEN + s1 + g * 16], 0, 1);
        }

        // ---- scores: two 16x16 f32 tiles, K-dim = ch chained over two WMMAs
        v8f S0 = {}, S1 = {};
        {
            v16h kB;
            kB = load_kb(klds[cur], m, 0, g);       S0 = wmma_f16(qA0, kB, S0);
            kB = load_kb(klds[cur], m, 32, g);      S0 = wmma_f16(qA1, kB, S0);
            kB = load_kb(klds[cur], 16 + m, 0, g);  S1 = wmma_f16(qA0, kB, S1);
            kB = load_kb(klds[cur], 16 + m, 32, g); S1 = wmma_f16(qA1, kB, S1);
        }

        // ---- W multiply + online softmax (D layout: row = r + 8*g, col = m)
        float Sw0[8], Sw1[8], rmax[8];
#pragma unroll
        for (int r = 0; r < 8; ++r) {
            size_t wrow = (size_t)(t0 + r + 8 * g) * T_LEN + s0;
            Sw0[r] = S0[r] * Wb[wrow + m];
            Sw1[r] = S1[r] * Wb[wrow + 16 + m];
            float v2 = fmaxf(Sw0[r], Sw1[r]);
#pragma unroll
            for (int msk = 8; msk >= 1; msk >>= 1)
                v2 = fmaxf(v2, __shfl_xor(v2, msk, 16));
            rmax[r] = v2;
        }
#pragma unroll
        for (int r = 0; r < 8; ++r) {
            float mnew  = fmaxf(Mrow[r], rmax[r]);
            float alpha = __expf(Mrow[r] - mnew);
            float p0 = __expf(Sw0[r] - mnew);
            float p1 = __expf(Sw1[r] - mnew);
            float rs = p0 + p1;
#pragma unroll
            for (int msk = 8; msk >= 1; msk >>= 1)
                rs += __shfl_xor(rs, msk, 16);
            Lrow[r] = Lrow[r] * alpha + rs;
            Mrow[r] = mnew;
            O0[r] *= alpha; O1[r] *= alpha; O2[r] *= alpha; O3[r] *= alpha;
            plds[wave][r + 8 * g][m]      = (_Float16)p0;
            plds[wave][r + 8 * g][16 + m] = (_Float16)p1;
        }
        // cross-lane LDS round-trip within the wave: D-layout -> A-layout
        asm volatile("s_wait_dscnt 0x0" ::: "memory");

        v16h pA;
#pragma unroll
        for (int v = 0; v < 8; ++v) {
            int k0 = a_kbase(v, g);
            pA[2 * v]     = plds[wave][m][k0];
            pA[2 * v + 1] = plds[wave][m][k0 + 1];
        }

        // ---- O += P * V^T  (K-dim = s = 32, one WMMA per 16-wide c chunk)
        {
            v16h vB;
            vB = load_vb(vlds[cur], 0, m, g); O0 = wmma_f16(pA, vB, O0);
            vB = load_vb(vlds[cur], 1, m, g); O1 = wmma_f16(pA, vB, O1);
            vB = load_vb(vlds[cur], 2, m, g); O2 = wmma_f16(pA, vB, O2);
            vB = load_vb(vlds[cur], 3, m, g); O3 = wmma_f16(pA, vB, O3);
        }
        __syncthreads();  // everyone done with buf[cur]; next buffer fully staged
    }

    // ---- normalize and store: out[c][t], D layout row t = t0 + r + 8g, col c
#pragma unroll
    for (int r = 0; r < 8; ++r) {
        float linv = 1.0f / Lrow[r];
        int t = t0 + r + 8 * g;
        ob[(size_t)(m) * T_LEN + t]      = O0[r] * linv;
        ob[(size_t)(16 + m) * T_LEN + t] = O1[r] * linv;
        ob[(size_t)(32 + m) * T_LEN + t] = O2[r] * linv;
        ob[(size_t)(48 + m) * T_LEN + t] = O3[r] * linv;
    }
}

extern "C" void kernel_launch(void* const* d_in, const int* in_sizes, int n_in,
                              void* d_out, int out_size, void* d_ws, size_t ws_size,
                              hipStream_t stream) {
    const float* qkv = (const float*)d_in[0];
    const float* W   = (const float*)d_in[1];
    float* out       = (float*)d_out;
    dim3 grid(T_LEN / BLOCK_T, NBH);
    wqkv_attn_kernel<<<grid, NTHREADS, 0, stream>>>(qkv, W, out);
}
